// get_model_42537356099786
// MI455X (gfx1250) — compile-verified, matrix operand order checked by reading, not run on hardware
//
#include <hip/hip_runtime.h>

// ---------------------------------------------------------------------------
// PointCNN forward for gfx1250 (MI455X, wave32, WMMA).
// All weight-shared matmuls -> v_wmma_f32_16x16x32_f16 (f16 in, f32 accum),
// staged through double-buffered LDS; blocks compute 128 x (16*NT) output
// (NT in {1,2,4} by N) so the A panel is read once per 64 output columns.
// Interior tiles take an unconditional, coalesced staging path; edges
// zero-pad. Fused bias + ReLU + folded-BN epilogue.  Irregular ops (dilated
// KNN, gathers, per-point X@fc, depthwise) on VALU.
// ---------------------------------------------------------------------------

typedef __attribute__((ext_vector_type(16))) _Float16 v16h;
typedef __attribute__((ext_vector_type(2)))  _Float16 h2;
typedef __attribute__((ext_vector_type(8)))  float    v8f;

#define BATCH  64
#define NPTS0  1024
#define NCLASS 40

// ---------------------------------------------------------------------------
// Generic GEMM:  C[m,n] = epi( sum_k A[m,k]*B[k,n] + bias[n] )
// A: (M,K) f32 row stride lda.  B: (K,N) f32 row stride N.
// Block: 256 threads = 8 waves; block computes 128(M) x 16*NT(N); wave w owns
// rows [blk*128 + 16w, +16) across all NT column tiles (shared A fragment,
// NT accumulators).  Double-buffered LDS, one barrier per 32-wide K step.
// ---------------------------------------------------------------------------
template <int NT>
__global__ void k_gemm_wmma(const float* __restrict__ A, int lda,
                            const float* __restrict__ Bm,
                            const float* __restrict__ bias,
                            const float* __restrict__ scale,
                            const float* __restrict__ shift,
                            float* __restrict__ C, int ldc,
                            int M, int N, int K, int relu)
{
    __shared__ _Float16 As[2][128][32];       // [buf][row][k]      16 KB
    __shared__ _Float16 Bs[2][16 * NT][32];   // [buf][n][k]  2*NT KB (B^T)

    int tid  = threadIdx.x;
    int wave = tid >> 5;
    int lane = tid & 31;
    int ntg  = (N + 16 * NT - 1) / (16 * NT);
    int bm   = (int)blockIdx.x / ntg;
    int bg   = (int)blockIdx.x % ntg;
    int row_blk  = bm << 7;
    int col_base = bg * (16 * NT);

    int half = lane >> 4;                 // 0: K 0-7/16-23, 1: K 8-15/24-31
    int l15  = lane & 15;
    int arow = (wave << 4) + l15;         // this lane's A row within strip

    // staging slots: A -> (r0 + 8*it, kta); B -> (j*16 + nb, ktb / ktb+16)
    int r0  = tid >> 5;
    int kta = tid & 31;
    int nb  = tid & 15;
    int ktb = tid >> 4;

    bool mfull = (row_blk + 128) <= M;
    bool nfull = (col_base + 16 * NT) <= N;

    int nsteps = (K + 31) >> 5;
    v8f acc[NT];
#pragma unroll
    for (int j = 0; j < NT; ++j) acc[j] = (v8f){};

    for (int s = 0; s < nsteps; ++s) {
        int k0  = s << 5;
        int buf = s & 1;
        bool kfull = (k0 + 32) <= K;

        if (mfull && kfull) {
            // -------- fast path: unconditional, coalesced, batched waits ----
            const float* Ab = A + (size_t)(row_blk + r0) * lda + (k0 + kta);
            float va[16];
#pragma unroll
            for (int it = 0; it < 16; ++it)
                va[it] = Ab[(size_t)it * 8 * lda];
#pragma unroll
            for (int it = 0; it < 16; ++it)
                As[buf][r0 + it * 8][kta] = (_Float16)va[it];
        } else {
            // -------- edge path: zero-padded ----
#pragma unroll
            for (int it = 0; it < 16; ++it) {
                int gr = row_blk + r0 + it * 8;
                int gk = k0 + kta;
                float v = (gr < M && gk < K) ? A[(size_t)gr * lda + gk] : 0.0f;
                As[buf][r0 + it * 8][kta] = (_Float16)v;
            }
        }

        if (nfull && kfull) {
#pragma unroll
            for (int j = 0; j < NT; ++j) {
                int gn = col_base + j * 16 + nb;
                float v0 = Bm[(size_t)(k0 + ktb) * N + gn];
                float v1 = Bm[(size_t)(k0 + ktb + 16) * N + gn];
                Bs[buf][j * 16 + nb][ktb]      = (_Float16)v0;
                Bs[buf][j * 16 + nb][ktb + 16] = (_Float16)v1;
            }
        } else {
#pragma unroll
            for (int j = 0; j < NT; ++j) {
                int gn = col_base + j * 16 + nb;
                int gk0 = k0 + ktb, gk1 = k0 + ktb + 16;
                float v0 = (gn < N && gk0 < K) ? Bm[(size_t)gk0 * N + gn] : 0.0f;
                float v1 = (gn < N && gk1 < K) ? Bm[(size_t)gk1 * N + gn] : 0.0f;
                Bs[buf][j * 16 + nb][ktb]      = (_Float16)v0;
                Bs[buf][j * 16 + nb][ktb + 16] = (_Float16)v1;
            }
        }
        __syncthreads();

        // ---- unconditional fragment loads from LDS (merge to ds_load_b128) --
        v16h af;
        {
            const h2* Arow = (const h2*)(&As[buf][arow][0]);
#pragma unroll
            for (int v = 0; v < 8; ++v) {
                int kk = ((v < 4) ? (2 * v) : (16 + 2 * (v - 4))) + 8 * half;
                h2 pa = Arow[kk >> 1];
                af[2 * v] = pa[0]; af[2 * v + 1] = pa[1];
            }
        }
#pragma unroll
        for (int j = 0; j < NT; ++j) {
            v16h bf;
            const h2* Brow = (const h2*)(&Bs[buf][j * 16 + l15][0]);
#pragma unroll
            for (int v = 0; v < 8; ++v) {
                int kk = ((v < 4) ? (2 * v) : (16 + 2 * (v - 4))) + 8 * half;
                h2 pb = Brow[kk >> 1];
                bf[2 * v] = pb[0]; bf[2 * v + 1] = pb[1];
            }
            acc[j] = __builtin_amdgcn_wmma_f32_16x16x32_f16(
                         false, af, false, bf, (short)0, acc[j], false, false);
        }
        // no trailing barrier: next step writes the other LDS buffer; this
        // buffer is only rewritten after the *next* barrier (step s+2).
    }

    // ---- epilogue: bias + ReLU + per-column scale/shift (folded BN) ----
#pragma unroll
    for (int j = 0; j < NT; ++j) {
        int n = col_base + j * 16 + l15;
        if (n < N) {
            float bs = bias  ? bias[n]  : 0.0f;
            float sc = scale ? scale[n] : 1.0f;
            float sh = shift ? shift[n] : 0.0f;
#pragma unroll
            for (int r = 0; r < 8; ++r) {
                int m = row_blk + (wave << 4) + r + 8 * half;
                if (m < M) {
                    float v = acc[j][r] + bs;
                    if (relu) v = v > 0.0f ? v : 0.0f;
                    C[(size_t)m * ldc + n] = v * sc + sh;
                }
            }
        }
    }
}

// ---------------------------------------------------------------------------
// (B,3,N) -> (B,N,3)
// ---------------------------------------------------------------------------
__global__ void k_transpose_x(const float* __restrict__ x, float* __restrict__ pts,
                              int Bn, int Nn)
{
    int t = blockIdx.x * blockDim.x + threadIdx.x;
    if (t >= Bn * Nn * 3) return;
    int d = t % 3; int bn = t / 3; int n = bn % Nn; int b = bn / Nn;
    pts[t] = x[((size_t)b * 3 + d) * Nn + n];
}

// rep = pts[:, sel_idx, :]
__global__ void k_gather_rep(const float* __restrict__ pts, const int* __restrict__ sel,
                             float* __restrict__ rep, int Bn, int Nn, int Pn)
{
    int t = blockIdx.x * blockDim.x + threadIdx.x;
    if (t >= Bn * Pn * 3) return;
    int d = t % 3; int bp = t / 3; int p = bp % Pn; int b = bp / Pn;
    rep[t] = pts[((size_t)b * Nn + sel[p]) * 3 + d];
}

// ---------------------------------------------------------------------------
// Dilated KNN: per (b,p) keep K*D+1 smallest distances (stable on ties, like
// top_k), emit every D-th index starting at 1 (drop self/nearest).
// ---------------------------------------------------------------------------
#define MAXLK 73
__global__ void k_knn(const float* __restrict__ rep, const float* __restrict__ pts,
                      int* __restrict__ idx, int Bn, int Pn, int Nn, int K, int D)
{
    int t = blockIdx.x * blockDim.x + threadIdx.x;
    if (t >= Bn * Pn) return;
    int b = t / Pn, p = t % Pn;
    const float* rp = rep + ((size_t)b * Pn + p) * 3;
    float rx = rp[0], ry = rp[1], rz = rp[2];
    int LK = K * D + 1;
    float dlist[MAXLK]; int ilist[MAXLK];
    for (int i = 0; i < LK; ++i) { dlist[i] = 3.4e38f; ilist[i] = 0; }
    const float* pb = pts + (size_t)b * Nn * 3;
    for (int n = 0; n < Nn; ++n) {
        __builtin_prefetch(pb + 3 * (n + 16), 0, 0);
        float dx = pb[3 * n] - rx, dy = pb[3 * n + 1] - ry, dz = pb[3 * n + 2] - rz;
        float d = dx * dx + dy * dy + dz * dz;
        if (d < dlist[LK - 1]) {
            int j = LK - 1;
            while (j > 0 && dlist[j - 1] > d) {
                dlist[j] = dlist[j - 1]; ilist[j] = ilist[j - 1]; --j;
            }
            dlist[j] = d; ilist[j] = n;
        }
    }
    int* op = idx + ((size_t)b * Pn + p) * K;
    for (int q = 0; q < K; ++q) op[q] = ilist[1 + q * D];
}

// ---------------------------------------------------------------------------
// Gather: pl[b,p,k,:] = pts[idx]-rep ; fc[b,p,k, Cmid + c] = f[idx, c]
// (fc rows have length Cc = Cmid + Chalf; fl will be GEMM'd into cols [0,Cmid))
// ---------------------------------------------------------------------------
__global__ void k_gather(const float* __restrict__ pts, const float* __restrict__ rep,
                         const float* __restrict__ f, const int* __restrict__ idx,
                         float* __restrict__ pl, float* __restrict__ fc,
                         int Bn, int Pn, int Nn, int K, int Chalf, int Cmid)
{
    int Cc = Chalf + Cmid;
    int t = blockIdx.x * blockDim.x + threadIdx.x;
    if (t >= Bn * Pn * K) return;
    int bp = t / K; int b = bp / Pn;
    int j = idx[t];
    const float* src = pts + ((size_t)b * Nn + j) * 3;
    const float* r = rep + (size_t)bp * 3;
    float* plo = pl + (size_t)t * 3;
    plo[0] = src[0] - r[0]; plo[1] = src[1] - r[1]; plo[2] = src[2] - r[2];
    const float* fs = f + ((size_t)b * Nn + j) * Chalf;
    float* fco = fc + (size_t)t * Cc + Cmid;
    for (int c = 0; c < Chalf; ++c) fco[c] = fs[c];
}

// fX[bp,i,c] = sum_j X[bp,i,j] * fc[bp,j,c]   (per-point KxK @ KxCc)
__global__ void k_fx(const float* __restrict__ X, const float* __restrict__ fc,
                     float* __restrict__ fX, int total, int K, int Cc)
{
    int per = K * Cc;
    int t = blockIdx.x * blockDim.x + threadIdx.x;
    if (t >= total * per) return;
    int bp = t / per; int rem = t % per; int i = rem / Cc; int c = rem % Cc;
    const float* Xr  = X  + (size_t)bp * K * K + (size_t)i * K;
    const float* fcr = fc + (size_t)bp * K * Cc + c;
    float s = 0.0f;
    for (int j = 0; j < K; ++j) s += Xr[j] * fcr[(size_t)j * Cc];
    fX[t] = s;
}

// y[bp, c*dm+m] = sum_k fX[bp,k,c]*dwW[c,m,k] + dwb[c*dm+m]
__global__ void k_dw(const float* __restrict__ fX, const float* __restrict__ dwW,
                     const float* __restrict__ dwb, float* __restrict__ y,
                     int total, int K, int Cc, int dm)
{
    int per = Cc * dm;
    int t = blockIdx.x * blockDim.x + threadIdx.x;
    if (t >= total * per) return;
    int bp = t / per; int rem = t % per; int c = rem / dm; int m = rem % dm;
    const float* fr = fX + (size_t)bp * K * Cc + c;
    const float* w = dwW + ((size_t)c * dm + m) * K;
    float s = dwb[rem];
    for (int k = 0; k < K; ++k) s += fr[(size_t)k * Cc] * w[k];
    y[t] = s;
}

// xt_conv_W (KK,3,K) -> Wt[(k*3+d)*KK + o]  so x0 = pl_flat(B*P,3K) @ Wt
__global__ void k_prep_xtw(const float* __restrict__ W, float* __restrict__ Wt,
                           int K, int KK)
{
    int t = blockIdx.x * blockDim.x + threadIdx.x;
    if (t >= KK * 3 * K) return;
    int k = t % K; int od = t / K; int d = od % 3; int o = od / 3;
    Wt[((size_t)k * 3 + d) * KK + o] = W[t];
}

// W (rows,cols) -> Wt (cols,rows)
__global__ void k_transpose_mat(const float* __restrict__ W, float* __restrict__ Wt,
                                int rows, int cols)
{
    int t = blockIdx.x * blockDim.x + threadIdx.x;
    if (t >= rows * cols) return;
    int c = t % cols; int r = t / cols;
    Wt[(size_t)c * rows + r] = W[t];
}

// Fold eval BatchNorm: scale = g/sqrt(v+eps), shift = beta - mean*scale
__global__ void k_bn_prep(const float* __restrict__ g, const float* __restrict__ be,
                          const float* __restrict__ mn, const float* __restrict__ vr,
                          float* __restrict__ scale, float* __restrict__ shift, int C)
{
    int c = blockIdx.x * blockDim.x + threadIdx.x;
    if (c >= C) return;
    float s = g[c] * rsqrtf(vr[c] + 1e-5f);
    scale[c] = s;
    shift[c] = be[c] - mn[c] * s;
}

// out[b,c] = mean_p logits[b,p,c]
__global__ void k_mean(const float* __restrict__ logits, float* __restrict__ out,
                       int Bn, int Pn, int C)
{
    int t = blockIdx.x * blockDim.x + threadIdx.x;
    if (t >= Bn * C) return;
    int c = t % C; int b = t / C;
    const float* lp = logits + (size_t)b * Pn * C + c;
    float s = 0.0f;
    for (int p = 0; p < Pn; ++p) s += lp[(size_t)p * C];
    out[t] = s * (1.0f / (float)Pn);
}

// ---------------------------------------------------------------------------
static inline int cdiv(int a, int b) { return (a + b - 1) / b; }

static void gemm(hipStream_t s, const float* A, int lda, const float* B,
                 const float* bias, const float* scale, const float* shift,
                 float* C, int ldc, int M, int N, int K, int relu)
{
    int mb = cdiv(M, 128);
    if (N > 32) {
        k_gemm_wmma<4><<<mb * cdiv(N, 64), 256, 0, s>>>(A, lda, B, bias, scale,
                                                        shift, C, ldc, M, N, K, relu);
    } else if (N > 16) {
        k_gemm_wmma<2><<<mb * cdiv(N, 32), 256, 0, s>>>(A, lda, B, bias, scale,
                                                        shift, C, ldc, M, N, K, relu);
    } else {
        k_gemm_wmma<1><<<mb * cdiv(N, 16), 256, 0, s>>>(A, lda, B, bias, scale,
                                                        shift, C, ldc, M, N, K, relu);
    }
}

extern "C" void kernel_launch(void* const* d_in, const int* in_sizes, int n_in,
                              void* d_out, int out_size, void* d_ws, size_t ws_size,
                              hipStream_t stream)
{
    (void)in_sizes; (void)n_in; (void)out_size; (void)ws_size;
    const float* x  = (const float*)d_in[0];
    const int*  sel = (const int*)d_in[1];

    // 19 params per layer in dict order
    const float* LPm[5][19];
    for (int l = 0; l < 5; ++l)
        for (int j = 0; j < 19; ++j)
            LPm[l][j] = (const float*)d_in[2 + l * 19 + j];
    const float* W1 = (const float*)d_in[97];  const float* b1 = (const float*)d_in[98];
    const float* W2 = (const float*)d_in[99];  const float* b2 = (const float*)d_in[100];
    const float* W3 = (const float*)d_in[101]; const float* b3 = (const float*)d_in[102];

    // -------- workspace carve ----------
    char* ws = (char*)d_ws; size_t off = 0;
    auto carve = [&](size_t elems) -> float* {
        float* p = (float*)(ws + off);
        off += ((elems * sizeof(float)) + 255) & ~(size_t)255;
        return p;
    };
    float* pts0 = carve((size_t)BATCH * NPTS0 * 3);
    float* repb = carve((size_t)BATCH * 120 * 3);
    float* ftsA = carve((size_t)BATCH * NPTS0 * 160);
    float* ftsB = carve((size_t)BATCH * NPTS0 * 160);
    float* fbuf = carve((size_t)BATCH * NPTS0 * 80);
    int*   idxb = (int*)carve((size_t)BATCH * NPTS0 * 8);
    float* plb  = carve((size_t)BATCH * NPTS0 * 8 * 3);
    float* tmp1 = carve((size_t)BATCH * NPTS0 * 8 * 24);
    float* fcb  = carve((size_t)BATCH * NPTS0 * 8 * 72);
    float* fXb  = carve((size_t)BATCH * NPTS0 * 8 * 72);
    float* x0b  = carve((size_t)BATCH * NPTS0 * 64);
    float* x1b  = carve((size_t)BATCH * NPTS0 * 64);
    float* yb   = carve((size_t)BATCH * NPTS0 * 144);
    float* xtwT = carve((size_t)36 * 144);
    float* pwT  = carve((size_t)240 * 160);
    float* bnsc = carve(160);
    float* bnsh = carve(160);

    // -------- input transpose: pts = fts0 = x^T ----------
    {
        int tot = BATCH * NPTS0 * 3;
        k_transpose_x<<<cdiv(tot, 256), 256, 0, stream>>>(x, pts0, BATCH, NPTS0);
    }

    const int cfgCin[5]  = {3, 32, 64, 96, 128};
    const int cfgCout[5] = {32, 64, 96, 128, 160};
    const int cfgK[5]    = {8, 8, 8, 12, 12};
    const int cfgD[5]    = {1, 2, 4, 4, 6};
    const int cfgP[5]    = {-1, -1, -1, 120, 120};

    float* pts_cur = pts0;
    const float* fts_cur = pts0;   // layer0 features == coords
    int cur_N = NPTS0;
    float* fts_bufs[2] = {ftsA, ftsB};
    int fb = 0;

    for (int l = 0; l < 5; ++l) {
        int Cin = cfgCin[l], Cout = cfgCout[l], K = cfgK[l], D = cfgD[l];
        int Chalf = Cout / 2, Cmid = Cout / 4, Cc = Chalf + Cmid;
        int dm = (Cout + Cin - 1) / Cin; if (dm > 4) dm = 4;
        int Ccdm = Cc * dm, KK = K * K;
        bool use_sel = (cfgP[l] > 0) && (cfgP[l] < cur_N);
        int P = use_sel ? cfgP[l] : cur_N;
        const float* const* p = LPm[l];

        const float* rep;
        if (use_sel) {
            int tot = BATCH * P * 3;
            k_gather_rep<<<cdiv(tot, 256), 256, 0, stream>>>(pts_cur, sel, repb,
                                                             BATCH, cur_N, P);
            rep = repb;
        } else {
            rep = pts_cur;
        }

        // f = relu(fts @ dense_W + b)   (B*cur_N, Cin) -> (B*cur_N, Chalf)
        gemm(stream, fts_cur, Cin, p[0], p[1], nullptr, nullptr,
             fbuf, Chalf, BATCH * cur_N, Chalf, Cin, 1);

        // dilated KNN indices
        k_knn<<<cdiv(BATCH * P, 128), 128, 0, stream>>>(rep, pts_cur, idxb,
                                                        BATCH, P, cur_N, K, D);

        // gather local coords + features into fc[:, Cmid:]
        k_gather<<<cdiv(BATCH * P * K, 256), 256, 0, stream>>>(
            pts_cur, rep, fbuf, idxb, plb, fcb, BATCH, P, cur_N, K, Chalf, Cmid);

        // lift pl -> fl (two dense+relu), fl written into fc[:, 0:Cmid]
        gemm(stream, plb, 3, p[2], p[3], nullptr, nullptr,
             tmp1, Cmid, BATCH * P * K, Cmid, 3, 1);
        gemm(stream, tmp1, Cmid, p[4], p[5], nullptr, nullptr,
             fcb, Cc, BATCH * P * K, Cmid, Cmid, 1);

        // X-transform
        k_prep_xtw<<<cdiv(KK * 3 * K, 256), 256, 0, stream>>>(p[6], xtwT, K, KK);
        gemm(stream, plb, 3 * K, xtwT, p[7], nullptr, nullptr,
             x0b, KK, BATCH * P, KK, 3 * K, 1);                    // x0 = relu(conv)
        gemm(stream, x0b, KK, p[8], p[9], nullptr, nullptr,
             x1b, KK, BATCH * P, KK, KK, 1);                       // x1 = relu(dense)
        gemm(stream, x1b, KK, p[10], p[11], nullptr, nullptr,
             x0b, KK, BATCH * P, KK, KK, 0);                       // X (no act)

        // fX = X @ fc  (per point)
        k_fx<<<cdiv(BATCH * P * K * Cc, 256), 256, 0, stream>>>(
            x0b, fcb, fXb, BATCH * P, K, Cc);

        // depthwise over K
        k_dw<<<cdiv(BATCH * P * Ccdm, 256), 256, 0, stream>>>(
            fXb, p[12], p[13], yb, BATCH * P, K, Cc, dm);

        // pointwise conv + ReLU + folded BN
        k_transpose_mat<<<cdiv(Cout * Ccdm, 256), 256, 0, stream>>>(p[14], pwT,
                                                                    Cout, Ccdm);
        k_bn_prep<<<cdiv(Cout, 128), 128, 0, stream>>>(p[15], p[16], p[17], p[18],
                                                       bnsc, bnsh, Cout);
        float* fts_next = fts_bufs[fb]; fb ^= 1;
        gemm(stream, yb, Ccdm, pwT, nullptr, bnsc, bnsh,
             fts_next, Cout, BATCH * P, Cout, Ccdm, 1);

        if (use_sel) pts_cur = repb;
        fts_cur = fts_next;
        cur_N = P;
    }

    // -------- FCN head + mean over points ----------
    int MP = BATCH * cur_N;  // cur_N == 120
    gemm(stream, fts_cur, 160, W1, b1, nullptr, nullptr, fbuf, 128, MP, 128, 160, 1);
    gemm(stream, fbuf, 128, W2, b2, nullptr, nullptr, tmp1, 64, MP, 64, 128, 1);
    gemm(stream, tmp1, 64, W3, b3, nullptr, nullptr, x0b, NCLASS, MP, NCLASS, 64, 0);
    k_mean<<<cdiv(BATCH * NCLASS, 128), 128, 0, stream>>>(x0b, (float*)d_out,
                                                          BATCH, cur_N, NCLASS);
}
